// LMDecoder_71777493451434
// MI455X (gfx1250) — matrix operand order repaired and checked by reading, not compiled, not run on hardware
//
#include <hip/hip_runtime.h>
#include <cstdint>
#include <cstddef>

// Problem dims (fixed by the reference)
#define V_SZ 32000
#define E_SZ 256
#define H_SZ 256
#define ROWS 8192      // B*T = 64*128

// GEMM tiling
#define TM 128         // M rows per workgroup
#define TN 128         // N cols per workgroup
#define KD 256         // full K depth
#define KH 128         // K half (double-buffered stage)

typedef _Float16 f16;
typedef _Float16 v16h __attribute__((ext_vector_type(16)));
typedef _Float16 v8h  __attribute__((ext_vector_type(8)));
typedef _Float16 v4h  __attribute__((ext_vector_type(4)));
typedef float    v8f  __attribute__((ext_vector_type(8)));

union AB16 { v16h v; v8h h[2]; };

__device__ __forceinline__ float sigmoidf_(float x) {
    return 1.0f / (1.0f + __expf(-x));
}

// ---------------------------------------------------------------------------
// Kernel A: embedding gather + gi = x*w_ih^T + b_ih + GRU gate math -> h(f16)
// One block per (b,t) row; embedding row staged in LDS; thread j computes
// gate components j, 256+j, 512+j.  3.2 GFLOP total: pure VALU, negligible.
// ---------------------------------------------------------------------------
__global__ __launch_bounds__(256)
void gru_h_f16(const int* __restrict__ tok, const float* __restrict__ emb,
               const float* __restrict__ w_ih, const float* __restrict__ b_ih,
               const float* __restrict__ b_hh, f16* __restrict__ h_out)
{
    __shared__ float xs[E_SZ];
    const int row = blockIdx.x;
    const int tid = threadIdx.x;
    const int t   = tok[row];
    xs[tid] = emb[(size_t)t * E_SZ + tid];
    __syncthreads();

    const float* wr = w_ih + (size_t)(tid)            * E_SZ;
    const float* wz = w_ih + (size_t)(H_SZ   + tid)   * E_SZ;
    const float* wn = w_ih + (size_t)(2*H_SZ + tid)   * E_SZ;

    float ar = 0.f, az = 0.f, an = 0.f;
#pragma unroll 8
    for (int k = 0; k < E_SZ; ++k) {
        const float x = xs[k];
        ar = fmaf(x, wr[k], ar);
        az = fmaf(x, wz[k], az);
        an = fmaf(x, wn[k], an);
    }
    const float r = sigmoidf_(ar + b_ih[tid]          + b_hh[tid]);
    const float z = sigmoidf_(az + b_ih[H_SZ + tid]   + b_hh[H_SZ + tid]);
    const float n = tanhf(    an + b_ih[2*H_SZ + tid] + r * b_hh[2*H_SZ + tid]);
    h_out[(size_t)row * H_SZ + tid] = (f16)((1.0f - z) * n);
}

// ---------------------------------------------------------------------------
// Kernel W: one-time w_cls fp32 -> f16 (48 MB traffic, ~2 us). Halves all
// downstream per-tile staging traffic and enables pure-copy async staging.
// ---------------------------------------------------------------------------
__global__ __launch_bounds__(256)
void wcls_to_f16(const float* __restrict__ src, f16* __restrict__ dst)
{
    const int i = blockIdx.x * 256 + threadIdx.x;       // float4 index
    const float4 f = ((const float4*)src)[i];
    v4h h = { (f16)f.x, (f16)f.y, (f16)f.z, (f16)f.w };
    ((v4h*)dst)[i] = h;
}

// ---------------------------------------------------------------------------
// Kernel B: logits[8192,32000] = h(f16) x w_cls_f16^T + b_cls
// WG = 8 waves, tile 128M x 128N. K=256 split in two 128-deep halves,
// double-buffered in LDS (4 x 32 KB = 128 KB) and filled with
// global_load_async_to_lds_b128 (ASYNCcnt pipeline):
//   issue half0 + half1 -> s_wait_asynccnt 16 -> compute half0
//                       -> s_wait_asynccnt 0  -> compute half1
// Waves 4(M) x 2(N), each 32x64 = 2x4 v_wmma_f32_16x16x32_f16 tiles.
// Epilogue stages the fp32 tile through LDS for coalesced b128 stores.
// ---------------------------------------------------------------------------
__global__ __launch_bounds__(256)
void cls_gemm_wmma(const f16* __restrict__ hmat, const f16* __restrict__ wclsh,
                   const float* __restrict__ b_cls, float* __restrict__ out)
{
    extern __shared__ char smem[];
    f16* Asb[2] = { (f16*)(smem),         (f16*)(smem + 65536) };
    f16* Bsb[2] = { (f16*)(smem + 32768), (f16*)(smem + 98304) };

    const int tid = threadIdx.x;
    const int m0  = blockIdx.y * TM;
    const int n0  = blockIdx.x * TN;

    // ---- issue async staging for both K-halves (16 instrs/thread/half) ----
#pragma unroll
    for (int ph = 0; ph < 2; ++ph) {
        const int kh = ph * KH;
        const unsigned ldsA = (unsigned)(uintptr_t)Asb[ph];
        const unsigned ldsB = (unsigned)(uintptr_t)Bsb[ph];
        // A half: 128 rows x 128 halves = 2048 b128 -> 8 per thread
#pragma unroll
        for (int j = 0; j < 8; ++j) {
            const int idx = tid + j * 256;
            const int r = idx >> 4;            // 16 b128 per row
            const int c = idx & 15;
            const unsigned goff = (unsigned)(((m0 + r) * KD + kh + c * 8) * 2);
            const unsigned loff = ldsA + (unsigned)((r * KH + c * 8) * 2);
            asm volatile("global_load_async_to_lds_b128 %0, %1, %2"
                         :: "v"(loff), "v"(goff), "s"(hmat) : "memory");
        }
        // B half: 128 rows x 128 halves = 2048 b128 -> 8 per thread
#pragma unroll
        for (int j = 0; j < 8; ++j) {
            const int idx = tid + j * 256;
            const int r = idx >> 4;
            const int c = idx & 15;
            const unsigned goff = (unsigned)(((n0 + r) * KD + kh + c * 8) * 2);
            const unsigned loff = ldsB + (unsigned)((r * KH + c * 8) * 2);
            asm volatile("global_load_async_to_lds_b128 %0, %1, %2"
                         :: "v"(loff), "v"(goff), "s"(wclsh) : "memory");
        }
    }

    const int wave   = tid >> 5;     // wave32
    const int lane   = tid & 31;
    const int laneLo = lane & 15;
    const int laneHi = lane >> 4;
    const int wm     = wave >> 1;    // 0..3 -> M quadrant (32 rows each)
    const int wn     = wave & 1;     // 0..1 -> N half (64 cols each)

    v8f acc[2][4] = {};

    // ---- K pipeline: compute half ph while (ph==0) half 1 is in flight ----
#pragma unroll
    for (int ph = 0; ph < 2; ++ph) {
        if (ph == 0) {
            asm volatile("s_wait_asynccnt 0x10" ::: "memory");   // half0 done
        } else {
            asm volatile("s_wait_asynccnt 0x0" ::: "memory");    // half1 done
        }
        __syncthreads();

        const f16* As = Asb[ph];
        const f16* Bs = Bsb[ph];
#pragma unroll
        for (int k0 = 0; k0 < KH; k0 += 32) {
            AB16 a[2], b[4];
            // A frag (16-bit 16x32): lanes 0-15: K k0..+7 / k0+16..+23,
            // lanes 16-31: K k0+8..+15 / k0+24..+31
#pragma unroll
            for (int mi = 0; mi < 2; ++mi) {
                const int m = wm * 32 + mi * 16 + laneLo;
                const f16* p = As + m * KH + k0 + laneHi * 8;
                a[mi].h[0] = *(const v8h*)(p);
                a[mi].h[1] = *(const v8h*)(p + 16);
            }
            // B frag (32x16): lane = N; lanes 0-15: K k0..+7 / k0+16..+23,
            // lanes 16-31: K k0+8..+15 / k0+24..+31
#pragma unroll
            for (int ni = 0; ni < 4; ++ni) {
                const int n = wn * 64 + ni * 16 + laneLo;
                const f16* p = Bs + n * KH + k0 + laneHi * 8;
                b[ni].h[0] = *(const v8h*)(p);
                b[ni].h[1] = *(const v8h*)(p + 16);
            }
#pragma unroll
            for (int mi = 0; mi < 2; ++mi)
#pragma unroll
                for (int ni = 0; ni < 4; ++ni)
                    acc[mi][ni] = __builtin_amdgcn_wmma_f32_16x16x32_f16(
                        false, a[mi].v, false, b[ni].v,
                        (short)0, acc[mi][ni], false, false);
        }
    }

    // ---- epilogue: stage fp32 tile in LDS, stream out as coalesced b128 ----
    __syncthreads();                      // everyone done reading f16 buffers
    float* Co = (float*)smem;             // 128x128 fp32 = 64 KB (reuse)
#pragma unroll
    for (int mi = 0; mi < 2; ++mi) {
#pragma unroll
        for (int ni = 0; ni < 4; ++ni) {
            const int col_l = wn * 64 + ni * 16 + laneLo;
            const int row_b = wm * 32 + mi * 16 + laneHi * 8;
#pragma unroll
            for (int v = 0; v < 8; ++v)
                Co[(row_b + v) * TN + col_l] = acc[mi][ni][v];
        }
    }
    __syncthreads();

    const int c4 = tid & 31;              // float4 column (constant per thread)
    const int r0 = tid >> 5;
    const float4 bb = ((const float4*)(b_cls + n0))[c4];
#pragma unroll
    for (int j = 0; j < 16; ++j) {
        const int r = r0 + j * 8;
        float4 vv = ((const float4*)Co)[r * (TN / 4) + c4];
        vv.x += bb.x; vv.y += bb.y; vv.z += bb.z; vv.w += bb.w;
        ((float4*)(out + (size_t)(m0 + r) * V_SZ + n0))[c4] = vv;
    }
}

// ---------------------------------------------------------------------------
extern "C" void kernel_launch(void* const* d_in, const int* in_sizes, int n_in,
                              void* d_out, int out_size, void* d_ws, size_t ws_size,
                              hipStream_t stream)
{
    const int*   tok   = (const int*)  d_in[0];   // target_sequence [64,128]
    const float* emb   = (const float*)d_in[1];   // [32000,256]
    const float* w_ih  = (const float*)d_in[2];   // [768,256]
    const float* b_ih  = (const float*)d_in[3];   // [768]
    const float* b_hh  = (const float*)d_in[4];   // [768]
    const float* w_cls = (const float*)d_in[5];   // [32000,256]
    const float* b_cls = (const float*)d_in[6];   // [32000]
    float* out = (float*)d_out;                   // [8192,32000] fp32

    f16* hbuf = (f16*)d_ws;                                   // 4 MB
    f16* wbuf = (f16*)((char*)d_ws + (size_t)ROWS * H_SZ * 2); // 16 MB

    gru_h_f16<<<ROWS, 256, 0, stream>>>(tok, emb, w_ih, b_ih, b_hh, hbuf);
    wcls_to_f16<<<(V_SZ * H_SZ / 4) / 256, 256, 0, stream>>>(w_cls, wbuf);

    dim3 grid(V_SZ / TN, ROWS / TM);              // 250 x 64
    const size_t lds = 4u * 32768u;               // 128 KB double-buffered
    cls_gemm_wmma<<<grid, 256, lds, stream>>>(hbuf, wbuf, b_cls, out);
}
